// Decoder_2791728742886
// MI455X (gfx1250) — matrix operand order repaired and checked by reading, not compiled, not run on hardware
//
#include <hip/hip_runtime.h>
#include <hip/hip_bf16.h>
#include <math.h>

// Problem sizes (fixed by the reference)
#define B_    32
#define H_    1024
#define FOURH 4096
#define TWOH  2048
#define T_    100
#define V_    32000
#define MBLK  8      // M-tiles per wave in the FC GEMM (128 rows / block)

typedef __bf16 bf16;
typedef __attribute__((ext_vector_type(16))) __bf16 v16bf;
typedef __attribute__((ext_vector_type(8)))  __bf16 bf16x8;
typedef __attribute__((ext_vector_type(8)))  float  v8f;

// ---------------------------------------------------------------------------
// WMMA fragment helpers (CDNA5 16x16x32 bf16 layouts, wave32)
// A: 16x32 (MxK). Lanes 0-15: row=lane, K=0..7 in v0-3, K=16..23 in v4-7.
//                 Lanes 16-31: row=lane-16, K=8..15 / K=24..31.
// ---------------------------------------------------------------------------
static __device__ __forceinline__ v16bf load_a_frag(const bf16* __restrict__ base, int ld) {
  const int lane = threadIdx.x & 31;
  const int row  = lane & 15;
  const int koff = (lane >> 4) << 3;          // 0 or 8
  const bf16* p = base + row * ld + koff;
  bf16x8 lo = *reinterpret_cast<const bf16x8*>(p);       // K koff..koff+7
  bf16x8 hi = *reinterpret_cast<const bf16x8*>(p + 16);  // K koff+16..koff+23
  v16bf f;
#pragma unroll
  for (int i = 0; i < 8; ++i) { f[i] = lo[i]; f[i + 8] = hi[i]; }
  return f;
}

// B: 32x16 (KxN) taken from a row-major W[N][K] block => computes A @ W^T.
// Lanes 0-15: n=lane, K=0..7 / K=16..23. Lanes 16-31: n=lane-16, K=8..15 / K=24..31.
static __device__ __forceinline__ v16bf load_b_frag(const bf16* __restrict__ base, int ld) {
  const int lane = threadIdx.x & 31;
  const int n    = lane & 15;
  const int koff = (lane >> 4) << 3;          // 0 or 8
  const bf16* p = base + n * ld + koff;
  bf16x8 lo = *reinterpret_cast<const bf16x8*>(p);
  bf16x8 hi = *reinterpret_cast<const bf16x8*>(p + 16);
  v16bf f;
#pragma unroll
  for (int i = 0; i < 8; ++i) { f[i] = lo[i]; f[i + 8] = hi[i]; }
  return f;
}

// C/D f32 layout: lane L holds column n=L&15; VGPR i holds row i (+8 for L>=16).
static __device__ __forceinline__ int c_row(int i) {
  return (((threadIdx.x & 31) >> 4) << 3) + i;
}

// ---------------------------------------------------------------------------
// fp32 -> bf16 conversion
// ---------------------------------------------------------------------------
__global__ void f2bf_kernel(const float* __restrict__ src, bf16* __restrict__ dst, size_t n) {
  size_t i = (size_t)blockIdx.x * blockDim.x + threadIdx.x;
  if (i < n) dst[i] = (bf16)src[i];
}

// h0 -> bf16 state, c0 -> fp32 state (2*B*H = 65536 elements)
__global__ void init_state_kernel(const float* __restrict__ h0, const float* __restrict__ c0,
                                  bf16* __restrict__ h_bf, float* __restrict__ c) {
  int i = blockIdx.x * blockDim.x + threadIdx.x;
  if (i < 2 * B_ * H_) { h_bf[i] = (bf16)h0[i]; c[i] = c0[i]; }
}

// ---------------------------------------------------------------------------
// xproj[dir][B][4H] = x_bf @ W_ih[dir]^T + b_ih[dir] + b_hh[dir]
// grid: (FOURH/16/8, 1, 2), block 256. Each wave computes both M-tiles (M=32)
// so every B-fragment feeds two WMMAs.
// ---------------------------------------------------------------------------
__global__ __launch_bounds__(256) void xproj_gemm_kernel(
    const bf16* __restrict__ x_bf,        // [B][H]
    const bf16* __restrict__ Wih_f,       // [4H][H] bf16
    const bf16* __restrict__ Wih_b,
    const float* __restrict__ bih_f, const float* __restrict__ bhh_f,
    const float* __restrict__ bih_b, const float* __restrict__ bhh_b,
    float* __restrict__ xproj)            // [2][B][4H]
{
  const int wave = threadIdx.x >> 5;
  const int dir  = blockIdx.z;
  const int n0   = (blockIdx.x * 8 + wave) * 16;
  const bf16* W   = (dir == 0) ? Wih_f : Wih_b;
  const float* bi = (dir == 0) ? bih_f : bih_b;
  const float* bh = (dir == 0) ? bhh_f : bhh_b;

  const bf16* Bw = W + (size_t)n0 * H_;
  const int col  = n0 + (threadIdx.x & 15);
  const float bias = bi[col] + bh[col];

  v8f acc0, acc1;
#pragma unroll
  for (int i = 0; i < 8; ++i) { acc0[i] = bias; acc1[i] = bias; }

  for (int k0 = 0; k0 < H_; k0 += 32) {
    v16bf b  = load_b_frag(Bw + k0, H_);
    v16bf a0 = load_a_frag(x_bf + k0, H_);
    v16bf a1 = load_a_frag(x_bf + 16 * H_ + k0, H_);
    acc0 = __builtin_amdgcn_wmma_f32_16x16x32_bf16(false, a0, false, b, (short)0, acc0, false, false);
    acc1 = __builtin_amdgcn_wmma_f32_16x16x32_bf16(false, a1, false, b, (short)0, acc1, false, false);
  }

  float* O = xproj + (size_t)dir * B_ * FOURH;
#pragma unroll
  for (int i = 0; i < 8; ++i) {
    O[(size_t)(c_row(i)) * FOURH + col]      = acc0[i];
    O[(size_t)(16 + c_row(i)) * FOURH + col] = acc1[i];
  }
}

// ---------------------------------------------------------------------------
// gates[dir][B][4H] = xproj[dir] + h_bf[dir] @ W_hh[dir]^T    (one time step)
// grid: (FOURH/16/8, 1, 2), block 256. Both M-tiles per wave: W_hh streamed
// from L2 exactly once per step per direction (8 MB) on the serial path.
// ---------------------------------------------------------------------------
__global__ __launch_bounds__(256) void gates_gemm_kernel(
    const bf16* __restrict__ h_bf,        // [2][B][H]
    const bf16* __restrict__ Whh_f,       // [4H][H]
    const bf16* __restrict__ Whh_b,
    const float* __restrict__ xproj,      // [2][B][4H]
    float* __restrict__ g)                // [2][B][4H]
{
  const int wave = threadIdx.x >> 5;
  const int dir  = blockIdx.z;
  const int n0   = (blockIdx.x * 8 + wave) * 16;

  const bf16* A  = h_bf + (size_t)dir * B_ * H_;
  const bf16* Bw = ((dir == 0) ? Whh_f : Whh_b) + (size_t)n0 * H_;
  const float* X = xproj + (size_t)dir * B_ * FOURH;
  const int col  = n0 + (threadIdx.x & 15);

  v8f acc0, acc1;
#pragma unroll
  for (int i = 0; i < 8; ++i) {
    acc0[i] = X[(size_t)(c_row(i)) * FOURH + col];
    acc1[i] = X[(size_t)(16 + c_row(i)) * FOURH + col];
  }

  for (int k0 = 0; k0 < H_; k0 += 32) {
    v16bf b  = load_b_frag(Bw + k0, H_);
    v16bf a0 = load_a_frag(A + k0, H_);
    v16bf a1 = load_a_frag(A + 16 * H_ + k0, H_);
    acc0 = __builtin_amdgcn_wmma_f32_16x16x32_bf16(false, a0, false, b, (short)0, acc0, false, false);
    acc1 = __builtin_amdgcn_wmma_f32_16x16x32_bf16(false, a1, false, b, (short)0, acc1, false, false);
  }

  float* G = g + (size_t)dir * B_ * FOURH;
#pragma unroll
  for (int i = 0; i < 8; ++i) {
    G[(size_t)(c_row(i)) * FOURH + col]      = acc0[i];
    G[(size_t)(16 + c_row(i)) * FOURH + col] = acc1[i];
  }
}

// ---------------------------------------------------------------------------
// LSTM pointwise: c,h update + write bf16 h state and seq[b][t][dir*H + n]
// 2*B*H = 65536 threads
// ---------------------------------------------------------------------------
__global__ __launch_bounds__(256) void lstm_pointwise_kernel(
    const float* __restrict__ g,          // [2][B][4H]  (i,f,g,o)
    float* __restrict__ c,                // [2][B][H]
    bf16* __restrict__ h_bf,              // [2][B][H]
    bf16* __restrict__ seq,               // [B][T][2H]
    int t)
{
  int tid = blockIdx.x * blockDim.x + threadIdx.x;
  if (tid >= 2 * B_ * H_) return;
  int n   = tid & (H_ - 1);
  int b   = (tid >> 10) & (B_ - 1);
  int dir = tid >> 15;

  const float* gd = g + (size_t)dir * B_ * FOURH + (size_t)b * FOURH;
  float gi = gd[n];
  float gf = gd[H_ + n];
  float gg = gd[2 * H_ + n];
  float go = gd[3 * H_ + n];

  float si = 1.0f / (1.0f + expf(-gi));
  float sf = 1.0f / (1.0f + expf(-gf));
  float so = 1.0f / (1.0f + expf(-go));

  float cn = sf * c[tid] + si * tanhf(gg);
  float hn = so * tanhf(cn);

  c[tid]    = cn;
  h_bf[tid] = (bf16)hn;
  seq[((size_t)b * T_ + t) * TWOH + dir * H_ + n] = (bf16)hn;
}

// ---------------------------------------------------------------------------
// logits[B*T][V] = seq @ W_fc^T + b_fc   (419 GFLOP, the dominant GEMM)
// grid: (V/16/8, B*T/128), block 256. Each wave: 8 M-tiles x 1 N-tile, so
// every 1KB B-fragment from L2 feeds 8 WMMAs (~128 FLOP/L2-byte). The 128x32
// A-tile is staged once per K-step into LDS and shared by all 8 waves.
// ---------------------------------------------------------------------------
__global__ __launch_bounds__(256) void fc_gemm_kernel(
    const bf16* __restrict__ seq,         // [B*T][2H]
    const bf16* __restrict__ Wfc,         // [V][2H]
    const float* __restrict__ bias,       // [V]
    float* __restrict__ out)              // [B*T][V]
{
  __shared__ bf16 As[16 * MBLK][40];      // 128x32 tile, +8 pad keeps 16B align

  const int wave = threadIdx.x >> 5;
  const int m0   = blockIdx.y * (16 * MBLK);
  const int n0   = (blockIdx.x * 8 + wave) * 16;

  const bf16* A  = seq + (size_t)m0 * TWOH;
  const bf16* Bw = Wfc + (size_t)n0 * TWOH;
  const int  col = threadIdx.x & 15;

  v8f acc[MBLK];
  {
    const float bv = bias[n0 + col];
#pragma unroll
    for (int j = 0; j < MBLK; ++j)
#pragma unroll
      for (int i = 0; i < 8; ++i) acc[j][i] = bv;
  }

  // LDS staging: 128 rows x 32 cols = 4096 bf16 per K-step; 256 threads load
  // 16 elements (32B) each: thread -> row = tid>>1, half = (tid&1)*16.
  const int sr = threadIdx.x >> 1;        // 0..127
  const int sk = (threadIdx.x & 1) << 4;  // 0 or 16

  const int lane  = threadIdx.x & 31;
  const int arow  = lane & 15;
  const int akoff = (lane >> 4) << 3;     // 0 or 8

  for (int k0 = 0; k0 < TWOH; k0 += 32) {
    __syncthreads();
    {
      const uint4* gp = reinterpret_cast<const uint4*>(A + (size_t)sr * TWOH + k0 + sk);
      uint4 d0 = gp[0];
      uint4 d1 = gp[1];
      *reinterpret_cast<uint4*>(&As[sr][sk])     = d0;
      *reinterpret_cast<uint4*>(&As[sr][sk + 8]) = d1;
    }
    __syncthreads();

    if (k0 + 64 < TWOH)
      __builtin_prefetch(Bw + k0 + 64, 0, 1);   // global_prefetch_b8 on weights

    v16bf b = load_b_frag(Bw + k0, TWOH);
#pragma unroll
    for (int j = 0; j < MBLK; ++j) {
      v16bf a;
#pragma unroll
      for (int i = 0; i < 8; ++i) {
        a[i]     = As[j * 16 + arow][akoff + i];
        a[i + 8] = As[j * 16 + arow][akoff + 16 + i];
      }
      acc[j] = __builtin_amdgcn_wmma_f32_16x16x32_bf16(false, a, false, b, (short)0, acc[j], false, false);
    }
  }

#pragma unroll
  for (int j = 0; j < MBLK; ++j)
#pragma unroll
    for (int i = 0; i < 8; ++i)
      out[(size_t)(m0 + j * 16 + c_row(i)) * V_ + n0 + col] = acc[j][i];
}

// ---------------------------------------------------------------------------
// In-place log_softmax over V per row. One block (256 threads) per row.
// ---------------------------------------------------------------------------
__global__ __launch_bounds__(256) void logsoftmax_kernel(float* __restrict__ out) {
  __shared__ float red[256];
  float* p = out + (size_t)blockIdx.x * V_;

  float m = -INFINITY;
  for (int i = threadIdx.x; i < V_; i += 256) m = fmaxf(m, p[i]);
  red[threadIdx.x] = m;
  __syncthreads();
  for (int s = 128; s > 0; s >>= 1) {
    if (threadIdx.x < s) red[threadIdx.x] = fmaxf(red[threadIdx.x], red[threadIdx.x + s]);
    __syncthreads();
  }
  m = red[0];
  __syncthreads();

  float sum = 0.0f;
  for (int i = threadIdx.x; i < V_; i += 256) sum += expf(p[i] - m);
  red[threadIdx.x] = sum;
  __syncthreads();
  for (int s = 128; s > 0; s >>= 1) {
    if (threadIdx.x < s) red[threadIdx.x] += red[threadIdx.x + s];
    __syncthreads();
  }
  const float lse = logf(red[0]) + m;
  __syncthreads();

  for (int i = threadIdx.x; i < V_; i += 256) p[i] = p[i] - lse;
}

// ---------------------------------------------------------------------------
// Host launcher
// ---------------------------------------------------------------------------
extern "C" void kernel_launch(void* const* d_in, const int* in_sizes, int n_in,
                              void* d_out, int out_size, void* d_ws, size_t ws_size,
                              hipStream_t stream) {
  (void)in_sizes; (void)n_in; (void)out_size; (void)ws_size;

  const float* x     = (const float*)d_in[0];
  const float* h0    = (const float*)d_in[1];
  const float* c0    = (const float*)d_in[2];
  const float* Wih_f = (const float*)d_in[3];
  const float* Whh_f = (const float*)d_in[4];
  const float* bih_f = (const float*)d_in[5];
  const float* bhh_f = (const float*)d_in[6];
  const float* Wih_b = (const float*)d_in[7];
  const float* Whh_b = (const float*)d_in[8];
  const float* bih_b = (const float*)d_in[9];
  const float* bhh_b = (const float*)d_in[10];
  const float* Wfc   = (const float*)d_in[11];
  const float* bfc   = (const float*)d_in[12];
  float* out = (float*)d_out;

  // ---- workspace carve-up (≈180 MB) ----
  char* ws = (char*)d_ws;
  size_t off = 0;
  auto carve = [&](size_t bytes) -> char* {
    char* p = ws + off;
    off += (bytes + 255) & ~(size_t)255;
    return p;
  };
  bf16*  bWih_f = (bf16*) carve((size_t)FOURH * H_ * 2);
  bf16*  bWih_b = (bf16*) carve((size_t)FOURH * H_ * 2);
  bf16*  bWhh_f = (bf16*) carve((size_t)FOURH * H_ * 2);
  bf16*  bWhh_b = (bf16*) carve((size_t)FOURH * H_ * 2);
  bf16*  bWfc   = (bf16*) carve((size_t)V_ * TWOH * 2);
  bf16*  bx     = (bf16*) carve((size_t)B_ * H_ * 2);
  float* xproj  = (float*)carve((size_t)2 * B_ * FOURH * 4);
  float* gates  = (float*)carve((size_t)2 * B_ * FOURH * 4);
  bf16*  h_bf   = (bf16*) carve((size_t)2 * B_ * H_ * 2);
  float* cstate = (float*)carve((size_t)2 * B_ * H_ * 4);
  bf16*  seq    = (bf16*) carve((size_t)B_ * T_ * TWOH * 2);

  auto cvt = [&](const float* s, bf16* d, size_t n) {
    unsigned blocks = (unsigned)((n + 255) / 256);
    f2bf_kernel<<<blocks, 256, 0, stream>>>(s, d, n);
  };

  // 1) one-time bf16 conversions
  cvt(Wih_f, bWih_f, (size_t)FOURH * H_);
  cvt(Wih_b, bWih_b, (size_t)FOURH * H_);
  cvt(Whh_f, bWhh_f, (size_t)FOURH * H_);
  cvt(Whh_b, bWhh_b, (size_t)FOURH * H_);
  cvt(Wfc,   bWfc,   (size_t)V_ * TWOH);
  cvt(x,     bx,     (size_t)B_ * H_);
  init_state_kernel<<<(2 * B_ * H_) / 256, 256, 0, stream>>>(h0, c0, h_bf, cstate);

  // 2) input projection (both directions)
  xproj_gemm_kernel<<<dim3(FOURH / 16 / 8, 1, 2), 256, 0, stream>>>(
      bx, bWih_f, bWih_b, bih_f, bhh_f, bih_b, bhh_b, xproj);

  // 3) sequential recurrence: 100 steps x (WMMA gate GEMM + pointwise update)
  for (int t = 0; t < T_; ++t) {
    gates_gemm_kernel<<<dim3(FOURH / 16 / 8, 1, 2), 256, 0, stream>>>(
        h_bf, bWhh_f, bWhh_b, xproj, gates);
    lstm_pointwise_kernel<<<(2 * B_ * H_) / 256, 256, 0, stream>>>(
        gates, cstate, h_bf, seq, t);
  }

  // 4) FLOP-dominant FC GEMM: 8 M-tiles per wave, A staged via LDS
  fc_gemm_kernel<<<dim3(V_ / 16 / 8, (B_ * T_) / (16 * MBLK)), 256, 0, stream>>>(
      seq, bWfc, bfc, out);

  // 5) in-place log_softmax over V
  logsoftmax_kernel<<<B_ * T_, 256, 0, stream>>>(out);
}
